// PointMixerInterSetLayer_27831388078302
// MI455X (gfx1250) — matrix operand order repaired and checked by reading, not compiled, not run on hardware
//
#include <hip/hip_runtime.h>
#include <hip/hip_bf16.h>

typedef __attribute__((ext_vector_type(16))) _Float16 v16h;
typedef __attribute__((ext_vector_type(8)))  float    v8f;

#define NPTS   50000
#define KNN    16
#define NKROWS (NPTS * KNN)          // 800000
#define NTILES (NKROWS / 16)         // 50000 (exact)
#define CIN    128
#define CS     16
#define BN_EPS 1e-5f

// ---- workspace layout (float offsets; all array bases 16B-aligned) ----
// coef region: A3[3]@0, B3[3]@3, M[3][16]@8, biasS[16]@56
#define WS_COEF     0
#define WS_STATPART 128                        // 256 blocks * 16 floats
#define WS_ARR      8192
#define OFF_M       ((size_t)WS_ARR)
#define OFF_Z       ((size_t)WS_ARR + (size_t)NPTS * CS)
#define OFF_RES     ((size_t)WS_ARR + 2ull * NPTS * CS)
#define OFF_S       ((size_t)WS_ARR + 3ull * NPTS * CS)
#define OFF_V       ((size_t)WS_ARR + 3ull * NPTS * CS + (size_t)NKROWS * CS)

#define STAT_BLOCKS 256

// ---------------- init: zero m / z / res (b128 stores) ----------------
__global__ void k_init(float* __restrict__ ws) {
    size_t i = (size_t)blockIdx.x * blockDim.x + threadIdx.x;
    size_t total4 = (3ull * NPTS * CS) >> 2;
    if (i < total4) {
        float4 z = {0.0f, 0.0f, 0.0f, 0.0f};
        *(float4*)(ws + OFF_M + 4 * i) = z;
    }
}

// ---------------- stage-1 moment reduction over p_r ----------------
__global__ void k_stats(const float* __restrict__ p_r, float* __restrict__ ws) {
    __shared__ float red[256];
    float acc[9];
#pragma unroll
    for (int q = 0; q < 9; ++q) acc[q] = 0.0f;
    for (size_t r = (size_t)blockIdx.x * blockDim.x + threadIdx.x; r < (size_t)NKROWS;
         r += (size_t)gridDim.x * blockDim.x) {
        float p0 = p_r[3 * r + 0], p1 = p_r[3 * r + 1], p2 = p_r[3 * r + 2];
        acc[0] += p0;      acc[1] += p1;      acc[2] += p2;
        acc[3] += p0 * p0; acc[4] += p1 * p1; acc[5] += p2 * p2;
        acc[6] += p0 * p1; acc[7] += p0 * p2; acc[8] += p1 * p2;
    }
#pragma unroll
    for (int q = 0; q < 9; ++q) {
        red[threadIdx.x] = acc[q];
        __syncthreads();
        for (int off = 128; off > 0; off >>= 1) {
            if ((int)threadIdx.x < off) red[threadIdx.x] += red[threadIdx.x + off];
            __syncthreads();
        }
        if (threadIdx.x == 0) ws[WS_STATPART + (size_t)blockIdx.x * 16 + q] = red[0];
        __syncthreads();
    }
}

// ---------------- fold p-branch into 3x16 matrix + BN affine ----------------
__global__ void k_coef(const float* __restrict__ Wlin, const float* __restrict__ blin,
                       const float* __restrict__ Wp1, const float* __restrict__ gamma,
                       const float* __restrict__ beta, const float* __restrict__ Wp2,
                       const float* __restrict__ bp2, float* __restrict__ ws) {
    int t = threadIdx.x;
    if (t < 48) {                       // M[j][c] = sum_d Wlin[c, d] * Wp2[d, j]
        int j = t / 16, c = t % 16;
        float a = 0.0f;
        for (int d = 0; d < CIN; ++d) a += Wlin[c * 256 + d] * Wp2[d * 3 + j];
        ws[8 + j * 16 + c] = a;
    } else if (t < 64) {                // biasS[c] = b_lin[c] + sum_d Wlin[c,d]*b_p2[d]
        int c = t - 48;
        float a = blin[c];
        for (int d = 0; d < CIN; ++d) a += Wlin[c * 256 + d] * bp2[d];
        ws[56 + c] = a;
    }
    if (t == 0) {                       // BN affine from moments of p
        float s[9];
#pragma unroll
        for (int q = 0; q < 9; ++q) s[q] = 0.0f;
        for (int b = 0; b < STAT_BLOCKS; ++b)
#pragma unroll
            for (int q = 0; q < 9; ++q) s[q] += ws[WS_STATPART + (size_t)b * 16 + q];
        const float inv = 1.0f / (float)NKROWS;
        float mp[3] = {s[0] * inv, s[1] * inv, s[2] * inv};
        float E[3][3];
        E[0][0] = s[3] * inv; E[1][1] = s[4] * inv; E[2][2] = s[5] * inv;
        E[0][1] = E[1][0] = s[6] * inv;
        E[0][2] = E[2][0] = s[7] * inv;
        E[1][2] = E[2][1] = s[8] * inv;
        for (int j = 0; j < 3; ++j) {
            float mu = 0.0f, e2 = 0.0f;
            for (int a = 0; a < 3; ++a) {
                mu += Wp1[j * 3 + a] * mp[a];
                for (int b = 0; b < 3; ++b)
                    e2 += Wp1[j * 3 + a] * Wp1[j * 3 + b] * E[a][b];
            }
            float var = e2 - mu * mu;
            float A = gamma[j] * rsqrtf(var + BN_EPS);
            ws[j]     = A;
            ws[3 + j] = beta[j] - A * mu;
        }
    }
}

// ---------------- fused main pass: dual WMMA GEMM + p-branch + segment max ----
__global__ void __launch_bounds__(256)
k_main(const float* __restrict__ x_knn, const int* __restrict__ knn_idx,
       const float* __restrict__ p_r, const float* __restrict__ Wlin,
       const float* __restrict__ Wx, const float* __restrict__ bx,
       const float* __restrict__ Wp1, float* __restrict__ ws) {
    const int lane = threadIdx.x & 31;
    const int mrow = lane & 15;
    const int hi   = lane >> 4;

    // weight B-fragments (tile-invariant), B[k][c]=W[c][k]; lanes 0-15: K 0..7,16..23, 16-31: K 8..15,24..31
    v16h bS[4], bV[4];
#pragma unroll
    for (int t = 0; t < 4; ++t)
#pragma unroll
        for (int i = 0; i < 16; ++i) {
            int k = 32 * t + 8 * hi + (i & 7) + ((i >> 3) << 4);
            bS[t][i] = (_Float16)Wlin[mrow * 256 + 128 + k];   // x_knn half of W_lin
            bV[t][i] = (_Float16)Wx[mrow * 128 + k];
        }

    // per-lane (column = mrow) coefficients
    const float M0 = ws[8 + mrow], M1 = ws[8 + 16 + mrow], M2 = ws[8 + 32 + mrow];
    const float biasS = ws[56 + mrow];
    const float bxr = bx[mrow];
    const float A0 = ws[0], A1 = ws[1], A2 = ws[2];
    const float B0 = ws[3], B1 = ws[4], B2 = ws[5];
    const float W00 = Wp1[0], W01 = Wp1[1], W02 = Wp1[2];
    const float W10 = Wp1[3], W11 = Wp1[4], W12 = Wp1[5];
    const float W20 = Wp1[6], W21 = Wp1[7], W22 = Wp1[8];

    float* mA = ws + OFF_M;

    const int waveGlobal = blockIdx.x * (blockDim.x >> 5) + (threadIdx.x >> 5);
    const int nWaves = gridDim.x * (blockDim.x >> 5);

    for (int tile = waveGlobal; tile < NTILES; tile += nWaves) {
        const int base = tile * 16;
        const float* xr = x_knn + (size_t)(base + mrow) * CIN;
        if (tile + nWaves < NTILES)
            __builtin_prefetch(x_knn + (size_t)(base + nWaves * 16 + mrow) * CIN, 0, 1);

        // p-branch: hb for tile row 'mrow'
        const float* pp = p_r + (size_t)(base + mrow) * 3;
        float p0 = pp[0], p1 = pp[1], p2 = pp[2];
        float hb0 = fmaxf(A0 * (W00 * p0 + W01 * p1 + W02 * p2) + B0, 0.0f);
        float hb1 = fmaxf(A1 * (W10 * p0 + W11 * p1 + W12 * p2) + B1, 0.0f);
        float hb2 = fmaxf(A2 * (W20 * p0 + W21 * p1 + W22 * p2) + B2, 0.0f);
        int idxl = knn_idx[base + mrow];

        v8f accS = {};
        v8f accV = {};
#pragma unroll
        for (int t = 0; t < 4; ++t) {
            // A-fragment per ISA 16-bit 16x32 layout:
            // hi=0: VGPR0-3 <- K 32t+0..7,  VGPR4-7 <- K 32t+16..23
            // hi=1: VGPR0-3 <- K 32t+8..15, VGPR4-7 <- K 32t+24..31
            const float4* c1 = (const float4*)(xr + 32 * t + 8 * hi);
            const float4* c2 = (const float4*)(xr + 32 * t + 16 + 8 * hi);
            float4 f0 = c1[0], f1 = c1[1], f2 = c2[0], f3 = c2[1];
            v16h a;
            a[0]  = (_Float16)f0.x; a[1]  = (_Float16)f0.y;
            a[2]  = (_Float16)f0.z; a[3]  = (_Float16)f0.w;
            a[4]  = (_Float16)f1.x; a[5]  = (_Float16)f1.y;
            a[6]  = (_Float16)f1.z; a[7]  = (_Float16)f1.w;
            a[8]  = (_Float16)f2.x; a[9]  = (_Float16)f2.y;
            a[10] = (_Float16)f2.z; a[11] = (_Float16)f2.w;
            a[12] = (_Float16)f3.x; a[13] = (_Float16)f3.y;
            a[14] = (_Float16)f3.z; a[15] = (_Float16)f3.w;
            accS = __builtin_amdgcn_wmma_f32_16x16x32_f16(false, a, false, bS[t],
                                                          (short)0, accS, false, false);
            accV = __builtin_amdgcn_wmma_f32_16x16x32_f16(false, a, false, bV[t],
                                                          (short)0, accV, false, false);
        }

        // D layout: VGPR r holds row r (lanes 0-15) / row r+8 (lanes 16-31), col = lane&15
        // Stores: constant-stride from one precomputed pointer -> immediate offsets
        float* sp = ws + OFF_S + (size_t)(base + 8 * hi) * CS + mrow;
        float* vp = ws + OFF_V + (size_t)(base + 8 * hi) * CS + mrow;
#pragma unroll
        for (int r = 0; r < 8; ++r) {
            int rt = r + 8 * hi;
            float g0 = __shfl(hb0, rt, 32);
            float g1 = __shfl(hb1, rt, 32);
            float g2 = __shfl(hb2, rt, 32);
            int   seg = __shfl(idxl, rt, 32);
            float pc = g0 * M0 + g1 * M1 + g2 * M2 + biasS;
            float sv = fmaxf(accS[r] + pc, 0.0f);
            float vv = fmaxf(accV[r] + bxr, 0.0f);
            sp[r * CS] = sv;
            vp[r * CS] = vv;
            // s >= 0 => uint ordering == float ordering; m initialized to 0.0f
            atomicMax((unsigned int*)(mA + (size_t)seg * CS + mrow), __float_as_uint(sv));
        }
    }
}

// ---------------- e = exp(s - m[idx]); z += e   (float4 per thread) ----------
__global__ void k_exp(const int* __restrict__ knn_idx, float* __restrict__ ws) {
    size_t t = (size_t)blockIdx.x * blockDim.x + threadIdx.x;
    if (t >= ((size_t)NKROWS * CS) >> 2) return;
    size_t row = t >> 2;
    int cb = (int)(t & 3) * 4;
    int seg = knn_idx[row];
    float4 s4 = *(const float4*)(ws + OFF_S + row * CS + cb);
    float4 m4 = *(const float4*)(ws + OFF_M + (size_t)seg * CS + cb);
    float4 e4;
    e4.x = __expf(s4.x - m4.x);
    e4.y = __expf(s4.y - m4.y);
    e4.z = __expf(s4.z - m4.z);
    e4.w = __expf(s4.w - m4.w);
    *(float4*)(ws + OFF_S + row * CS + cb) = e4;   // overwrite s with e
    float* zp = ws + OFF_Z + (size_t)seg * CS + cb;
    atomicAdd(zp + 0, e4.x);
    atomicAdd(zp + 1, e4.y);
    atomicAdd(zp + 2, e4.z);
    atomicAdd(zp + 3, e4.w);
}

// ---------------- res += v * e / z[idx]   (float4 per thread) ----------------
__global__ void k_res(const int* __restrict__ knn_idx, float* __restrict__ ws) {
    size_t t = (size_t)blockIdx.x * blockDim.x + threadIdx.x;
    if (t >= ((size_t)NKROWS * CS) >> 2) return;
    size_t row = t >> 2;
    int cb = (int)(t & 3) * 4;
    int seg = knn_idx[row];
    float4 e4 = *(const float4*)(ws + OFF_S + row * CS + cb);
    float4 v4 = *(const float4*)(ws + OFF_V + row * CS + cb);
    float4 z4 = *(const float4*)(ws + OFF_Z + (size_t)seg * CS + cb);
    float* rp = ws + OFF_RES + (size_t)seg * CS + cb;
    atomicAdd(rp + 0, v4.x * (e4.x / z4.x));
    atomicAdd(rp + 1, v4.y * (e4.y / z4.y));
    atomicAdd(rp + 2, v4.z * (e4.z / z4.z));
    atomicAdd(rp + 3, v4.w * (e4.w / z4.w));
}

// ---------------- out = x + tile(res, 8)   (float4 per thread) ----------------
__global__ void k_out(const float* __restrict__ x, const float* __restrict__ ws,
                      float* __restrict__ out) {
    size_t t = (size_t)blockIdx.x * blockDim.x + threadIdx.x;
    if (t >= ((size_t)NPTS * CIN) >> 2) return;
    size_t n = t >> 5;                    // 32 float4 groups per point
    int cb = (int)(t & 31) * 4;           // channel base 0..124
    float4 xv = *(const float4*)(x + n * CIN + cb);
    float4 rv = *(const float4*)(ws + OFF_RES + n * CS + (cb & 15));
    float4 o;
    o.x = xv.x + rv.x;
    o.y = xv.y + rv.y;
    o.z = xv.z + rv.z;
    o.w = xv.w + rv.w;
    *(float4*)(out + n * CIN + cb) = o;
}

extern "C" void kernel_launch(void* const* d_in, const int* in_sizes, int n_in,
                              void* d_out, int out_size, void* d_ws, size_t ws_size,
                              hipStream_t stream) {
    const float* x       = (const float*)d_in[0];
    const float* x_knn   = (const float*)d_in[1];
    const int*   knn_idx = (const int*)d_in[2];
    const float* p_r     = (const float*)d_in[3];
    const float* W_lin   = (const float*)d_in[4];
    const float* b_lin   = (const float*)d_in[5];
    const float* W_x     = (const float*)d_in[6];
    const float* b_x     = (const float*)d_in[7];
    const float* W_p1    = (const float*)d_in[8];
    const float* bn_g    = (const float*)d_in[9];
    const float* bn_b    = (const float*)d_in[10];
    const float* W_p2    = (const float*)d_in[11];
    const float* b_p2    = (const float*)d_in[12];
    float* ws  = (float*)d_ws;
    float* out = (float*)d_out;

    {   // zero m / z / res
        size_t total4 = (3ull * NPTS * CS) >> 2;
        int blocks = (int)((total4 + 255) / 256);
        k_init<<<blocks, 256, 0, stream>>>(ws);
    }
    k_stats<<<STAT_BLOCKS, 256, 0, stream>>>(p_r, ws);
    k_coef<<<1, 64, 0, stream>>>(W_lin, b_lin, W_p1, bn_g, bn_b, W_p2, b_p2, ws);
    k_main<<<6250, 256, 0, stream>>>(x_knn, knn_idx, p_r, W_lin, W_x, b_x, W_p1, ws);
    {
        size_t total4 = ((size_t)NKROWS * CS) >> 2;
        int blocks = (int)((total4 + 255) / 256);
        k_exp<<<blocks, 256, 0, stream>>>(knn_idx, ws);
        k_res<<<blocks, 256, 0, stream>>>(knn_idx, ws);
    }
    {
        size_t total4 = ((size_t)NPTS * CIN) >> 2;
        int blocks = (int)((total4 + 255) / 256);
        k_out<<<blocks, 256, 0, stream>>>(x, ws, out);
    }
}
